// EncoderDecoderModelWithPairInput_39479339385201
// MI455X (gfx1250) — compile-verified, matrix operand order checked by reading, not run on hardware
//
#include <hip/hip_runtime.h>
#include <hip/hip_bf16.h>
#include <cstddef>

// ---------------------------------------------------------------------------
// CDNA5 (gfx1250) wave32 WMMA building blocks.
// All f16 data is "pair-packed": u32 = { f16 k, f16 k+1 }, matching the WMMA
// fragment register layout so fragment loads are ds_load_b32/b128 with no
// half-word repacking.
// ---------------------------------------------------------------------------
typedef __attribute__((ext_vector_type(16))) _Float16 v16h;
typedef __attribute__((ext_vector_type(8)))  float    v8f;

union FragH { v16h h; unsigned u[8]; };

__device__ __forceinline__ unsigned pk2(float a, float b) {
    union { _Float16 h[2]; unsigned u; } p;
    p.h[0] = (_Float16)a;
    p.h[1] = (_Float16)b;
    return p.u;
}

// pair index q = k/2 for fragment slot v, lane-half hi (ISA 7.12.2 layout:
// lanes 0-15 hold K {0..7,16..23}, lanes 16-31 hold K {8..15,24..31}).
__device__ __forceinline__ int frag_q(int v, int hi) {
    return ((v >> 2) << 3) + (hi << 2) + (v & 3);
}

__device__ __forceinline__ v16h frag_a_pk(const unsigned* sh, int ldq, int lane) {
    int m = lane & 15, hi = lane >> 4;
    FragH f;
#pragma unroll
    for (int v = 0; v < 8; ++v) f.u[v] = sh[m * ldq + frag_q(v, hi)];
    return f.h;
}

__device__ __forceinline__ v16h frag_b_pk(const unsigned* sh, int ldn, int cb, int lane) {
    int n = lane & 15, hi = lane >> 4;
    FragH f;
#pragma unroll
    for (int v = 0; v < 8; ++v) f.u[v] = sh[frag_q(v, hi) * ldn + cb + n];
    return f.h;
}

__device__ __forceinline__ v8f wmma_f16(v16h a, v16h b, v8f c) {
    return __builtin_amdgcn_wmma_f32_16x16x32_f16(false, a, false, b, (short)0, c, false, false);
}

// ---------------------------------------------------------------------------
// gfx1250 async global->LDS copies (ISA 15.18.3, ASYNCcnt). ASYNCcnt retires
// loads in order, so s_wait_asynccnt <N> with N = just-issued count acts as a
// "previous slice has landed" wait for double buffering.
// ---------------------------------------------------------------------------
__device__ __forceinline__ unsigned lds_off(const void* p) {
    return (unsigned)(unsigned long long)p;
}
__device__ __forceinline__ void async_b128(unsigned lds, const void* g) {
    asm volatile("global_load_async_to_lds_b128 %0, %1, off" :: "v"(lds), "v"(g) : "memory");
}
template <int N>
__device__ __forceinline__ void async_waitN() {
    asm volatile("s_wait_asynccnt %0" :: "i"(N) : "memory");
}

// ---------------------------------------------------------------------------
// Weight pre-packing: f32 [K][N] row-major -> pair-packed f16 u32 [K/2][ldp],
// zero-padded for n >= N so async tile loads are always in-bounds.
// ---------------------------------------------------------------------------
__global__ void pack_pairs(const float* __restrict__ W, unsigned* __restrict__ P,
                           int K, int N, int ldp) {
    int idx = blockIdx.x * 256 + threadIdx.x;
    int total = (K / 2) * ldp;
    if (idx >= total) return;
    int q = idx / ldp, n = idx - q * ldp;
    float a = 0.f, b = 0.f;
    if (n < N) {
        a = W[(size_t)(2 * q) * N + n];
        b = W[(size_t)(2 * q + 1) * N + n];
    }
    P[idx] = pk2(a, b);
}

// att_W [H,d,hid] -> pair-packed [d/2][d] with column c = h*hid + t.
__global__ void pack_attw(const float* __restrict__ attW, unsigned* __restrict__ P,
                          int H, int d, int hid) {
    int idx = blockIdx.x * 256 + threadIdx.x;
    int total = (d / 2) * d;
    if (idx >= total) return;
    int q = idx / d, c = idx - q * d;
    int h = c / hid, t = c - h * hid;
    float a = attW[((size_t)h * d + 2 * q) * hid + t];
    float b = attW[((size_t)h * d + 2 * q + 1) * hid + t];
    P[idx] = pk2(a, b);
}

// ---------------------------------------------------------------------------
// Generic WMMA GEMM:  C[M,N] = act(A[M,K] @ W[K,N] + bias)
// Block = 256 threads = 8 waves; tile 16(M) x 256(N); 2 WMMAs / wave / k-step.
// W tiles double-buffered via async global->LDS b128; A staged packed f16.
// ---------------------------------------------------------------------------
template <int ACT>  // 0 = none, 1 = relu
__global__ __launch_bounds__(256) void gemm_wmma(const float* __restrict__ A,
                                                 const unsigned* __restrict__ Wp,
                                                 const float* __restrict__ bias,
                                                 float* __restrict__ C,
                                                 int M, int N, int K, int ldp) {
    __shared__ __align__(16) unsigned Apk[2][16 * 16];   // 16 rows x 16 pair-cols
    __shared__ __align__(16) unsigned Wt[2][16 * 256];   // 16 pair-rows x 256 cols
    const int m0 = blockIdx.x * 16;
    const int n0 = blockIdx.y * 256;
    const int tid = threadIdx.x;
    const int lane = tid & 31, wave = tid >> 5;
    const int nk = K / 32;

    auto stageA = [&](int buf, int k0) {
        int m = tid >> 4, q = tid & 15;
        float ax = 0.f, ay = 0.f;
        if (m0 + m < M) {
            ax = A[(size_t)(m0 + m) * K + k0 + 2 * q];
            ay = A[(size_t)(m0 + m) * K + k0 + 2 * q + 1];
        }
        Apk[buf][m * 16 + q] = pk2(ax, ay);
    };
    auto issueW = [&](int buf, int k0) {   // 16 pair-rows x 256 u32 = 16KB, 64B/thread
        int pr = tid >> 4, ch = tid & 15;
        const unsigned* g = Wp + (size_t)(k0 / 2 + pr) * ldp + n0 + ch * 16;
        unsigned l = lds_off(&Wt[buf][pr * 256 + ch * 16]);
#pragma unroll
        for (int c = 0; c < 4; ++c) async_b128(l + 16u * c, g + 4 * c);
    };

    v8f acc[2] = {{}, {}};
    stageA(0, 0);
    issueW(0, 0);
    for (int k = 0; k < nk; ++k) {
        const int cur = k & 1, nxt = cur ^ 1;
        if (k + 1 < nk) {
            stageA(nxt, (k + 1) * 32);
            issueW(nxt, (k + 1) * 32);
            async_waitN<4>();            // 4 outstanding = next slice; current landed
        } else {
            async_waitN<0>();
        }
        __syncthreads();
        v16h a  = frag_a_pk(Apk[cur], 16, lane);
        v16h b0 = frag_b_pk(Wt[cur], 256, wave * 32, lane);
        v16h b1 = frag_b_pk(Wt[cur], 256, wave * 32 + 16, lane);
        acc[0] = wmma_f16(a, b0, acc[0]);
        acc[1] = wmma_f16(a, b1, acc[1]);
        __syncthreads();
    }
    const int nl = lane & 15, hi = lane >> 4;
#pragma unroll
    for (int t = 0; t < 2; ++t) {
        int n = n0 + wave * 32 + t * 16 + nl;
        if (n < N) {
            float bv = bias ? bias[n] : 0.f;
#pragma unroll
            for (int r = 0; r < 8; ++r) {
                int m = m0 + r + 8 * hi;
                if (m < M) {
                    float v = acc[t][r] + bv;
                    if (ACT == 1) v = fmaxf(v, 0.f);
                    C[(size_t)m * N + n] = v;
                }
            }
        }
    }
}

// ---------------------------------------------------------------------------
// Char-CNN: embed -> conv1d(k=5, pad=2) -> relu -> maxpool over word dim.
// ---------------------------------------------------------------------------
__global__ __launch_bounds__(128) void char_embed(const int* __restrict__ enc,
                                                  const int* __restrict__ dec,
                                                  const float* __restrict__ emb,
                                                  const float* __restrict__ cw,
                                                  const float* __restrict__ cb,
                                                  float* __restrict__ feats,
                                                  int Le, int Ld) {
    const int Wl = 16, E = 64, F = 128, Kc = 5;
    __shared__ float x[16][64];
    const int Nn = Le + Ld;
    const int bn = blockIdx.x;
    const int b = bn / Nn, n = bn - b * Nn;
    const int tid = threadIdx.x;
    for (int idx = tid; idx < Wl * E; idx += 128) {
        int w = idx >> 6, e = idx & 63;
        int id = (n < Le) ? enc[((size_t)b * Le + n) * Wl + w]
                          : dec[((size_t)b * Ld + (n - Le)) * Wl + w];
        x[w][e] = emb[(size_t)id * E + e];
    }
    __syncthreads();
    const int f = tid;
    float mx = 0.f;
    for (int p = 0; p < Wl; ++p) {
        float acc = cb[f];
#pragma unroll
        for (int k = 0; k < Kc; ++k) {
            int q = p + k - 2;
            if (q < 0 || q >= Wl) continue;
            const float* wf = cw + (size_t)f * E * Kc + k;
#pragma unroll 8
            for (int e = 0; e < E; ++e) acc += x[q][e] * wf[e * Kc];
        }
        mx = fmaxf(mx, fmaxf(acc, 0.f));
    }
    feats[(size_t)bn * F + f] = mx;
}

// ---------------------------------------------------------------------------
// Fused pairwise dynamic graph. One block per (b,i):
//   h1[j,e] = relu(hi[i,e] + hj[j,e] + b0[e])        (pair-packed f16 in LDS)
//   h2      = relu(h1 @ W1 + b1)                     (WMMA, never stored)
//   dyn[j]  = sigmoid(sum_n h2[j,n]*wout[n] + bout)  (fused epilogue)
//   adj[i,j]= clip(dyn + fixed, 0, 1)
// Nn=400 (25 j-tiles), d=256 (8 k-steps, 16 n-tiles). W1 slices async
// double-buffered; slice-0 copy overlaps the h1 tile build.
// ---------------------------------------------------------------------------
__global__ __launch_bounds__(256) void pairwise_graph(const float* __restrict__ hi_buf,
                                                      const float* __restrict__ hj_buf,
                                                      const float* __restrict__ b0,
                                                      const unsigned* __restrict__ W1p,
                                                      const float* __restrict__ b1,
                                                      const float* __restrict__ wout,
                                                      const float* __restrict__ bout,
                                                      const int* __restrict__ fixed,
                                                      float* __restrict__ adj,
                                                      int Nn, int d) {
    __shared__ float base[256];
    __shared__ float b1s[256];
    __shared__ float wouts[256];
    __shared__ float sums[400];
    __shared__ __align__(16) unsigned h1pk[16 * 128];   // 16 j-rows x 128 pair-cols
    __shared__ __align__(16) unsigned w1t[2][16 * 256]; // double-buffered slices
    const int bi = blockIdx.x;              // b*Nn + i
    const int b = bi / Nn;
    const int tid = threadIdx.x;
    const int lane = tid & 31, wave = tid >> 5;

    auto issueW1 = [&](int buf, int k0) {   // contiguous 16KB slice, 64B/thread
        const unsigned* g = W1p + (size_t)(k0 / 2) * 256 + tid * 16;
        unsigned l = lds_off(&w1t[buf][tid * 16]);
#pragma unroll
        for (int c = 0; c < 4; ++c) async_b128(l + 16u * c, g + 4 * c);
    };

    base[tid]  = hi_buf[(size_t)bi * d + tid] + b0[tid];
    b1s[tid]   = b1[tid];
    wouts[tid] = wout[tid];
    for (int j = tid; j < Nn; j += 256) sums[j] = 0.f;
    __syncthreads();

    for (int jt = 0; jt < Nn / 16; ++jt) {
        __syncthreads();
        issueW1(0, 0);                      // overlap with h1 build
        // build h1 tile pair-packed: 2048 u32, 8 / thread (coalesced hj loads)
#pragma unroll
        for (int e = 0; e < 8; ++e) {
            int idx = tid + e * 256;
            int r = idx >> 7, q = idx & 127;
            int j = jt * 16 + r;
            const float* hjrow = hj_buf + (size_t)(b * Nn + j) * d + 2 * q;
            float v0 = fmaxf(base[2 * q]     + hjrow[0], 0.f);
            float v1 = fmaxf(base[2 * q + 1] + hjrow[1], 0.f);
            h1pk[r * 128 + q] = pk2(v0, v1);
        }
        v8f acc0 = {}, acc1 = {};
        for (int k = 0; k < 8; ++k) {       // d/32 = 8 k-steps
            const int cur = k & 1, nxt = cur ^ 1;
            if (k + 1 < 8) {
                issueW1(nxt, (k + 1) * 32);
                async_waitN<4>();           // current slice landed
            } else {
                async_waitN<0>();
            }
            __syncthreads();
            v16h a   = frag_a_pk(h1pk + k * 16, 128, lane);
            v16h bb0 = frag_b_pk(w1t[cur], 256, wave * 32, lane);
            v16h bb1 = frag_b_pk(w1t[cur], 256, wave * 32 + 16, lane);
            acc0 = wmma_f16(a, bb0, acc0);
            acc1 = wmma_f16(a, bb1, acc1);
            __syncthreads();
        }
        const int nl = lane & 15, hi2 = lane >> 4;
#pragma unroll
        for (int r = 0; r < 8; ++r) {
            int jl = r + 8 * hi2;
            int n0 = wave * 32 + nl;
            int n1 = n0 + 16;
            float v0 = fmaxf(acc0[r] + b1s[n0], 0.f) * wouts[n0];
            float v1 = fmaxf(acc1[r] + b1s[n1], 0.f) * wouts[n1];
            atomicAdd(&sums[jt * 16 + jl], v0 + v1);   // ds_add_f32
        }
    }
    __syncthreads();
    const float bo = bout[0];
    for (int j = tid; j < Nn; j += 256) {
        float s = sums[j] + bo;
        float dyn = __builtin_amdgcn_rcpf(1.f + __expf(-s));
        float a = dyn + (float)fixed[(size_t)bi * Nn + j];
        adj[(size_t)bi * Nn + j] = fminf(fmaxf(a, 0.f), 1.f);
    }
}

// f1/f2: per-(b,h,n) dot of Wh row with attention vectors.
__global__ void gat_f(const float* __restrict__ Wh,
                      const float* __restrict__ a1, const float* __restrict__ a2,
                      float* __restrict__ f1, float* __restrict__ f2,
                      int B, int H, int hid, int Nn, int d) {
    int idx = blockIdx.x * 256 + threadIdx.x;
    int total = B * H * Nn;
    if (idx >= total) return;
    int b = idx / (H * Nn);
    int rem = idx - b * (H * Nn);
    int h = rem / Nn, n = rem - h * Nn;
    const float* row = Wh + ((size_t)b * Nn + n) * d + h * hid;
    float s1 = 0.f, s2 = 0.f;
#pragma unroll
    for (int t = 0; t < 32; ++t) {
        float w = row[t];
        s1 += w * a1[h * hid + t];
        s2 += w * a2[h * hid + t];
    }
    f1[idx] = s1;
    f2[idx] = s2;
}

// Masked softmax attention + aggregation. One block per (b,h,i).
__global__ __launch_bounds__(256) void gat_attn(const float* __restrict__ Wh,
                                                const float* __restrict__ f1,
                                                const float* __restrict__ f2,
                                                const float* __restrict__ adj,
                                                float* __restrict__ xout,
                                                int B, int H, int Nn, int hid) {
    __shared__ float attn[400];
    __shared__ float rbuf[256];
    __shared__ float p2[8][32];
    const int blk = blockIdx.x;
    const int b = blk / (H * Nn);
    const int rem = blk - b * (H * Nn);
    const int h = rem / Nn, i = rem - h * Nn;
    const int tid = threadIdx.x;
    const int d = H * hid;
    const float fi = f1[((size_t)b * H + h) * Nn + i];

    float lmax = -3.4e38f;
    for (int j = tid; j < Nn; j += 256) {
        float e = fi + f2[((size_t)b * H + h) * Nn + j];
        e = (e > 0.f) ? e : 0.2f * e;                       // leaky_relu(0.2)
        if (!(adj[((size_t)b * Nn + i) * Nn + j] > 0.f)) e = -9e15f;
        attn[j] = e;
        lmax = fmaxf(lmax, e);
    }
    rbuf[tid] = lmax;
    __syncthreads();
    for (int s = 128; s > 0; s >>= 1) {
        if (tid < s) rbuf[tid] = fmaxf(rbuf[tid], rbuf[tid + s]);
        __syncthreads();
    }
    const float gmax = rbuf[0];
    __syncthreads();
    float lsum = 0.f;
    for (int j = tid; j < Nn; j += 256) {
        float ev = __expf(attn[j] - gmax);
        attn[j] = ev;
        lsum += ev;
    }
    rbuf[tid] = lsum;
    __syncthreads();
    for (int s = 128; s > 0; s >>= 1) {
        if (tid < s) rbuf[tid] += rbuf[tid + s];
        __syncthreads();
    }
    const float inv = __builtin_amdgcn_rcpf(rbuf[0]);
    __syncthreads();

    const int t = tid & 31, g = tid >> 5;
    float pacc = 0.f;
    for (int j = g; j < Nn; j += 8)
        pacc += attn[j] * Wh[((size_t)b * Nn + j) * d + h * hid + t];
    p2[g][t] = pacc;
    __syncthreads();
    if (g == 0) {
        float s = 0.f;
#pragma unroll
        for (int gg = 0; gg < 8; ++gg) s += p2[gg][t];
        s *= inv;
        float o = (s > 0.f) ? s : (__expf(s) - 1.f);        // elu
        xout[((size_t)b * Nn + i) * d + h * hid + t] = o;
    }
}

// Copy decoder rows x[:, Le:, :] into a compact [B*dec, d] matrix.
__global__ void slice_dec(const float* __restrict__ x, float* __restrict__ Adec,
                          int Nn, int enc, int d, int B) {
    int idx = blockIdx.x * 256 + threadIdx.x;
    int dec = Nn - enc;
    int total = B * dec * d;
    if (idx >= total) return;
    int b = idx / (dec * d);
    int rem = idx - b * (dec * d);
    int r = rem / d, c = rem - r * d;
    Adec[idx] = x[((size_t)b * Nn + enc + r) * d + c];
}

// ---------------------------------------------------------------------------
// Host orchestration
// ---------------------------------------------------------------------------
extern "C" void kernel_launch(void* const* d_in, const int* in_sizes, int n_in,
                              void* d_out, int out_size, void* d_ws, size_t ws_size,
                              hipStream_t stream) {
    (void)in_sizes; (void)n_in; (void)out_size; (void)ws_size;
    const int* enc   = (const int*)d_in[0];
    const int* dec_i = (const int*)d_in[1];
    const int* fixed = (const int*)d_in[2];
    const float* emb   = (const float*)d_in[3];
    const float* convw = (const float*)d_in[4];
    const float* convb = (const float*)d_in[5];
    const float* hstW  = (const float*)d_in[6];
    const float* hstb  = (const float*)d_in[7];
    const float* W0a   = (const float*)d_in[8];
    const float* W0b   = (const float*)d_in[9];
    const float* b0    = (const float*)d_in[10];
    const float* W1    = (const float*)d_in[11];
    const float* b1    = (const float*)d_in[12];
    const float* Wout  = (const float*)d_in[13];
    const float* bout  = (const float*)d_in[14];
    const float* attW  = (const float*)d_in[15];
    const float* a1    = (const float*)d_in[16];
    const float* a2    = (const float*)d_in[17];
    const float* outW  = (const float*)d_in[18];
    const float* outb  = (const float*)d_in[19];
    float* out = (float*)d_out;

    const int B = 2, Le = 250, Ld = 150, Nn = 400;
    const int F = 128, H = 8, hid = 32, d = 256, Vout = 5000, dec = 150;
    const int BN = B * Nn;           // 800
    const int VoutP = 5120;          // Vout padded to 256 for async tiles

    char* ws = (char*)d_ws;
    auto alloc = [&ws](size_t bytes) -> void* {
        void* p = (void*)ws;
        ws += (bytes + 255) & ~(size_t)255;
        return p;
    };
    float*    feats = (float*)alloc((size_t)BN * F * 4);
    float*    xA    = (float*)alloc((size_t)BN * d * 4);
    float*    xB    = (float*)alloc((size_t)BN * d * 4);
    float*    hiB   = (float*)alloc((size_t)BN * d * 4);
    float*    hjB   = (float*)alloc((size_t)BN * d * 4);
    float*    adj   = (float*)alloc((size_t)B * Nn * Nn * 4);
    float*    Wh    = (float*)alloc((size_t)BN * d * 4);
    float*    f1    = (float*)alloc((size_t)B * H * Nn * 4);
    float*    f2    = (float*)alloc((size_t)B * H * Nn * 4);
    float*    Adec  = (float*)alloc((size_t)B * dec * d * 4);
    unsigned* hstWp = (unsigned*)alloc((size_t)(F / 2) * d * 4);
    unsigned* W0ap  = (unsigned*)alloc((size_t)(d / 2) * d * 4);
    unsigned* W0bp  = (unsigned*)alloc((size_t)(d / 2) * d * 4);
    unsigned* W1p   = (unsigned*)alloc((size_t)(d / 2) * d * 4);
    unsigned* attp  = (unsigned*)alloc((size_t)(d / 2) * d * 4);
    unsigned* outWp = (unsigned*)alloc((size_t)(d / 2) * VoutP * 4);

    // 0) pre-pack all weights to pair-packed f16
    pack_pairs<<<((F / 2) * d + 255) / 256, 256, 0, stream>>>(hstW, hstWp, F, d, d);
    pack_pairs<<<((d / 2) * d + 255) / 256, 256, 0, stream>>>(W0a, W0ap, d, d, d);
    pack_pairs<<<((d / 2) * d + 255) / 256, 256, 0, stream>>>(W0b, W0bp, d, d, d);
    pack_pairs<<<((d / 2) * d + 255) / 256, 256, 0, stream>>>(W1, W1p, d, d, d);
    pack_attw<<<((d / 2) * d + 255) / 256, 256, 0, stream>>>(attW, attp, H, d, hid);
    pack_pairs<<<((d / 2) * VoutP + 255) / 256, 256, 0, stream>>>(outW, outWp, d, Vout, VoutP);

    // 1) char-CNN features
    char_embed<<<BN, 128, 0, stream>>>(enc, dec_i, emb, convw, convb, feats, Le, Ld);
    // 2) x = feats @ hst_W + hst_b   (M=800, K=128, N=256)
    gemm_wmma<0><<<dim3(BN / 16, 1), 256, 0, stream>>>(feats, hstWp, hstb, xA, BN, d, F, d);

    float* xcur = xA;
    float* xnxt = xB;
    for (int it = 0; it < 2; ++it) {
        gemm_wmma<0><<<dim3(BN / 16, 1), 256, 0, stream>>>(xcur, W0ap, nullptr, hiB, BN, d, d, d);
        gemm_wmma<0><<<dim3(BN / 16, 1), 256, 0, stream>>>(xcur, W0bp, nullptr, hjB, BN, d, d, d);
        pairwise_graph<<<BN, 256, 0, stream>>>(hiB, hjB, b0, W1p, b1, Wout, bout, fixed, adj, Nn, d);
        gemm_wmma<0><<<dim3(BN / 16, 1), 256, 0, stream>>>(xcur, attp, nullptr, Wh, BN, d, d, d);
        gat_f<<<(B * H * Nn + 255) / 256, 256, 0, stream>>>(Wh, a1, a2, f1, f2, B, H, hid, Nn, d);
        gat_attn<<<B * H * Nn, 256, 0, stream>>>(Wh, f1, f2, adj, xnxt, B, H, Nn, hid);
        float* t = xcur; xcur = xnxt; xnxt = t;
    }

    // 4) output projection: x[:, 250:, :] @ out_W + out_b  (M=300, N=5000, K=256)
    slice_dec<<<(B * dec * d + 255) / 256, 256, 0, stream>>>(xcur, Adec, Nn, Le, d, B);
    gemm_wmma<0><<<dim3((B * dec + 15) / 16, VoutP / 256), 256, 0, stream>>>(
        Adec, outWp, outb, out, B * dec, Vout, d, VoutP);
}
